// Attention_14998025798522
// MI455X (gfx1250) — compile-verified, matrix operand order checked by reading, not run on hardware
//
#include <hip/hip_runtime.h>

#define DIMC  768
#define HEADS 12
#define HDIM  64
#define BATCH 8
#define SEQ   1024
#define QSCALE 0.125f   // HD^-0.5

typedef __attribute__((ext_vector_type(16))) _Float16 v16h;
typedef __attribute__((ext_vector_type(8)))  _Float16 v8h;
typedef __attribute__((ext_vector_type(8)))  float    v8f;

// Load a 16-bit A/B WMMA fragment for one K=32 step from a row-major row
// pointer. Per CDNA5 layout: lane<16 holds K {k0..k0+7, k0+16..k0+23} with
// k0=0, lane>=16 with k0=8. Two 16-byte chunks per lane.
__device__ __forceinline__ v16h frag_ld(const _Float16* __restrict__ row, int k0) {
  v8h lo = *(const v8h*)(row + k0);
  v8h hi = *(const v8h*)(row + k0 + 16);
  v16h r;
#pragma unroll
  for (int t = 0; t < 8; ++t) { r[t] = lo[t]; r[t + 8] = hi[t]; }
  return r;
}

__device__ __forceinline__ v8f wmma16(v16h a, v16h b, v8f c) {
  return __builtin_amdgcn_wmma_f32_16x16x32_f16(false, a, false, b,
                                                (short)0, c, false, false);
}

__device__ __forceinline__ v8f vzero8() {
  v8f z;
#pragma unroll
  for (int e = 0; e < 8; ++e) z[e] = 0.f;
  return z;
}

// ---------------------------------------------------------------- converts
__global__ void convert_x_kernel(const float* __restrict__ x,
                                 _Float16* __restrict__ xh, int n) {
  int i = blockIdx.x * blockDim.x + threadIdx.x;
  if (i < n) xh[i] = (_Float16)x[i];
}

__global__ void convert_w_kernel(const float* __restrict__ Wq,
                                 const float* __restrict__ Wk,
                                 const float* __restrict__ Wv,
                                 const float* __restrict__ Wp,
                                 _Float16* __restrict__ WqT,
                                 _Float16* __restrict__ WkT,
                                 _Float16* __restrict__ WvT,
                                 _Float16* __restrict__ WpT) {
  int idx = blockIdx.x * blockDim.x + threadIdx.x;
  if (idx >= DIMC * DIMC) return;
  int kk = idx / DIMC, nn = idx % DIMC;
  int tid = nn * DIMC + kk;              // transposed: [n][k]
  WqT[tid] = (_Float16)Wq[idx];
  WkT[tid] = (_Float16)Wk[idx];
  WvT[tid] = (_Float16)Wv[idx];
  WpT[tid] = (_Float16)Wp[idx];
}

// ---------------------------------------------------------------- QKV GEMM
// grid = (M/16, 9), block = 128 (4 waves). col groups 0-2 -> q, 3-5 -> k, 6-8 -> v.
// q scaled by QSCALE. v written TRANSPOSED per head: [b][h][d][j].
__global__ __launch_bounds__(128)
void qkv_gemm_kernel(const _Float16* __restrict__ xh,
                     const _Float16* __restrict__ WqT,
                     const _Float16* __restrict__ WkT,
                     const _Float16* __restrict__ WvT,
                     _Float16* __restrict__ qb, _Float16* __restrict__ kb,
                     _Float16* __restrict__ vb) {
  const int mt = blockIdx.x, cg = blockIdx.y;
  const int wave = threadIdx.x >> 5, ln = threadIdx.x & 31;
  const int hlf = ln >> 4, lc = ln & 15;
  const int m0 = mt * 16;
  const int which = cg / 3;  // 0=q 1=k 2=v
  const _Float16* WT = (which == 0) ? WqT : (which == 1) ? WkT : WvT;
  const int nbase = (cg % 3) * 256 + wave * 64;  // within [0,768)
  const int kfrag = hlf * 8;

  const _Float16* arow = xh + (size_t)(m0 + lc) * DIMC;
  const _Float16* brow[4];
#pragma unroll
  for (int t = 0; t < 4; ++t)
    brow[t] = WT + (size_t)(nbase + t * 16 + lc) * DIMC;

  v8f c[4];
#pragma unroll
  for (int t = 0; t < 4; ++t) c[t] = vzero8();

  // software-pipelined, fully unrolled k loop: fragments for step s+1 are in
  // flight while step s WMMAs execute; full unroll lets SSA renaming remove
  // all buffer-rotation register copies.
  v16h a = frag_ld(arow, kfrag);
  v16h bfr[4];
#pragma unroll
  for (int t = 0; t < 4; ++t) bfr[t] = frag_ld(brow[t], kfrag);

  const int NSTEP = DIMC / 32;
#pragma unroll
  for (int s = 0; s < NSTEP - 1; ++s) {
    const int ks2 = (s + 1) * 32 + kfrag;
    v16h a_n = frag_ld(arow, ks2);
    v16h b_n[4];
#pragma unroll
    for (int t = 0; t < 4; ++t) b_n[t] = frag_ld(brow[t], ks2);
#pragma unroll
    for (int t = 0; t < 4; ++t) c[t] = wmma16(a, bfr[t], c[t]);
    a = a_n;
#pragma unroll
    for (int t = 0; t < 4; ++t) bfr[t] = b_n[t];
  }
#pragma unroll
  for (int t = 0; t < 4; ++t) c[t] = wmma16(a, bfr[t], c[t]);

  const int b_ = m0 / SEQ;
  const int nrow0 = m0 % SEQ;
  if (which == 2) {
    // V: transposed store [b][h][d][j]
#pragma unroll
    for (int t = 0; t < 4; ++t) {
      int col = nbase + t * 16 + lc;
      int head = col >> 6, d = col & 63;
      _Float16* vt = vb + (((size_t)b_ * HEADS + head) * HDIM + d) * SEQ;
#pragma unroll
      for (int r = 0; r < 8; ++r)
        vt[nrow0 + r + hlf * 8] = (_Float16)c[t][r];
    }
  } else {
    const float scale = (which == 0) ? QSCALE : 1.f;
    _Float16* dst = (which == 0) ? qb : kb;
#pragma unroll
    for (int t = 0; t < 4; ++t) {
      int col = nbase + t * 16 + lc;
      int head = col >> 6, d = col & 63;
#pragma unroll
      for (int r = 0; r < 8; ++r) {
        int i = nrow0 + r + hlf * 8;
        dst[(((size_t)b_ * HEADS + head) * SEQ + i) * HDIM + d] =
            (_Float16)(c[t][r] * scale);
      }
    }
  }
}

// ---------------------------------------------------- fused talking-heads attn
// grid = (SEQ/16, BATCH), block = 384 (12 waves, wave g owns head g).
// q,k layout [b][h][j][d]; v layout [b][h][d][j] (pre-transposed).
__global__ __launch_bounds__(384)
void attn_kernel(const _Float16* __restrict__ q, const _Float16* __restrict__ k,
                 const _Float16* __restrict__ v,
                 const float* __restrict__ Wl, const float* __restrict__ bl,
                 const float* __restrict__ Ww, const float* __restrict__ bw,
                 _Float16* __restrict__ oh) {
  __shared__ __align__(16) float    Smix[HEADS][16][16];   // raw score exchange
  __shared__ __align__(16) _Float16 Pn[HEADS][16][32];     // normalized probs
  __shared__ __align__(16) _Float16 P2[HEADS][16][32];     // post-mixed probs
  __shared__ float sWl[HEADS][HEADS], sWw[HEADS][HEADS];
  __shared__ float sbl[HEADS], sbw[HEADS];

  const int b = blockIdx.y;
  const int i0 = blockIdx.x * 16;
  const int wv = threadIdx.x >> 5;  // head
  const int ln = threadIdx.x & 31;
  const int hlf = ln >> 4, lc = ln & 15;
  const int kfrag = hlf * 8;

  if (threadIdx.x < HEADS * HEADS) {
    (&sWl[0][0])[threadIdx.x] = Wl[threadIdx.x];
    (&sWw[0][0])[threadIdx.x] = Ww[threadIdx.x];
  }
  if (threadIdx.x < HEADS) { sbl[threadIdx.x] = bl[threadIdx.x];
                             sbw[threadIdx.x] = bw[threadIdx.x]; }
  __syncthreads();

  const size_t headBase = ((size_t)b * HEADS + wv) * SEQ;
  const _Float16* vtb = v + headBase * HDIM;  // [d][j] for this (b, head)
  const _Float16* qrow = q + (headBase + i0 + lc) * HDIM;
  v16h aq0 = frag_ld(qrow, kfrag);
  v16h aq1 = frag_ld(qrow, kfrag + 32);

  float m[8], l[8];
#pragma unroll
  for (int r = 0; r < 8; ++r) { m[r] = -1e30f; l[r] = 0.f; }

  // ---------------- pass 1: online softmax stats over mixed scores ----------
  for (int j0 = 0; j0 < SEQ; j0 += 16) {
    const _Float16* krow = k + (headBase + j0 + lc) * HDIM;
    if (j0 + 16 < SEQ)
      __builtin_prefetch(k + (headBase + j0 + 16 + lc) * HDIM, 0, 1);
    v16h bk0 = frag_ld(krow, kfrag);
    v16h bk1 = frag_ld(krow, kfrag + 32);
    v8f c = vzero8();
    c = wmma16(aq0, bk0, c);
    c = wmma16(aq1, bk1, c);
#pragma unroll
    for (int r = 0; r < 8; ++r) Smix[wv][r + hlf * 8][lc] = c[r];
    __syncthreads();
#pragma unroll
    for (int r = 0; r < 8; ++r) {
      float s2 = sbl[wv];
#pragma unroll
      for (int h = 0; h < HEADS; ++h) s2 += sWl[wv][h] * Smix[h][r + hlf * 8][lc];
      float mx = s2;  // row max across the 16 lanes of this half
      mx = fmaxf(mx, __shfl_xor(mx, 1, 32));
      mx = fmaxf(mx, __shfl_xor(mx, 2, 32));
      mx = fmaxf(mx, __shfl_xor(mx, 4, 32));
      mx = fmaxf(mx, __shfl_xor(mx, 8, 32));
      float nm = fmaxf(m[r], mx);
      float p = __expf(s2 - nm);
      p += __shfl_xor(p, 1, 32);
      p += __shfl_xor(p, 2, 32);
      p += __shfl_xor(p, 4, 32);
      p += __shfl_xor(p, 8, 32);
      l[r] = l[r] * __expf(m[r] - nm) + p;
      m[r] = nm;
    }
    __syncthreads();
  }

  float rl[8];
#pragma unroll
  for (int r = 0; r < 8; ++r) rl[r] = 1.f / l[r];

  v8f oacc[4];
#pragma unroll
  for (int t = 0; t < 4; ++t) oacc[t] = vzero8();

  // ---------------- pass 2: probs, post-mix, P @ V (j-steps of 32) ----------
  for (int j0 = 0; j0 < SEQ; j0 += 32) {
    if (j0 + 32 < SEQ) {
      __builtin_prefetch(k + (headBase + j0 + 32 + lc) * HDIM, 0, 1);
      __builtin_prefetch(vtb + (size_t)(lc * 4) * SEQ + j0 + 32, 0, 1);
    }
    // two 16-col subtiles: scores -> pre-mix -> normalized probs
#pragma unroll
    for (int sub = 0; sub < 2; ++sub) {
      const int jb = j0 + sub * 16;
      const _Float16* krow = k + (headBase + jb + lc) * HDIM;
      v16h bk0 = frag_ld(krow, kfrag);
      v16h bk1 = frag_ld(krow, kfrag + 32);
      v8f c = vzero8();
      c = wmma16(aq0, bk0, c);
      c = wmma16(aq1, bk1, c);
#pragma unroll
      for (int r = 0; r < 8; ++r) Smix[wv][r + hlf * 8][lc] = c[r];
      __syncthreads();
#pragma unroll
      for (int r = 0; r < 8; ++r) {
        float s2 = sbl[wv];
#pragma unroll
        for (int h = 0; h < HEADS; ++h)
          s2 += sWl[wv][h] * Smix[h][r + hlf * 8][lc];
        Pn[wv][r + hlf * 8][sub * 16 + lc] = (_Float16)(__expf(s2 - m[r]) * rl[r]);
      }
      __syncthreads();
    }
    // post-softmax mix over heads for all 32 cols
    // (+bw per element: summed over all 1024 j reproduces the bias exactly)
#pragma unroll
    for (int sub = 0; sub < 2; ++sub) {
#pragma unroll
      for (int r = 0; r < 8; ++r) {
        float p2 = sbw[wv];
#pragma unroll
        for (int h = 0; h < HEADS; ++h)
          p2 += sWw[wv][h] * (float)Pn[h][r + hlf * 8][sub * 16 + lc];
        P2[wv][r + hlf * 8][sub * 16 + lc] = (_Float16)p2;
      }
    }
    __syncthreads();
    // A = P2[wv] (16x32), B = V^T tiles straight from global: full K=32 WMMAs
    v16h a = frag_ld(&P2[wv][lc][0], kfrag);
#pragma unroll
    for (int t = 0; t < 4; ++t) {
      const int d = t * 16 + lc;
      v16h bb = frag_ld(vtb + (size_t)d * SEQ + j0, kfrag);
      oacc[t] = wmma16(a, bb, oacc[t]);
    }
    __syncthreads();
  }

  // write (B, N, DIM) f16
#pragma unroll
  for (int t = 0; t < 4; ++t) {
    int col = wv * HDIM + t * 16 + lc;
#pragma unroll
    for (int r = 0; r < 8; ++r) {
      int i = i0 + r + hlf * 8;
      oh[((size_t)b * SEQ + i) * DIMC + col] = (_Float16)oacc[t][r];
    }
  }
}

// ---------------------------------------------------------------- out proj
// grid = (M/16, 3), block = 128 (4 waves x 64 cols).
__global__ __launch_bounds__(128)
void out_proj_kernel(const _Float16* __restrict__ oh,
                     const _Float16* __restrict__ WpT,
                     const float* __restrict__ bp, float* __restrict__ out) {
  const int mt = blockIdx.x, cg = blockIdx.y;
  const int wave = threadIdx.x >> 5, ln = threadIdx.x & 31;
  const int hlf = ln >> 4, lc = ln & 15;
  const int m0 = mt * 16;
  const int nbase = cg * 256 + wave * 64;
  const int kfrag = hlf * 8;

  const _Float16* arow = oh + (size_t)(m0 + lc) * DIMC;
  const _Float16* brow[4];
#pragma unroll
  for (int t = 0; t < 4; ++t)
    brow[t] = WpT + (size_t)(nbase + t * 16 + lc) * DIMC;

  v8f c[4];
#pragma unroll
  for (int t = 0; t < 4; ++t) c[t] = vzero8();

  v16h a = frag_ld(arow, kfrag);
  v16h bfr[4];
#pragma unroll
  for (int t = 0; t < 4; ++t) bfr[t] = frag_ld(brow[t], kfrag);

  const int NSTEP = DIMC / 32;
#pragma unroll
  for (int s = 0; s < NSTEP - 1; ++s) {
    const int ks2 = (s + 1) * 32 + kfrag;
    v16h a_n = frag_ld(arow, ks2);
    v16h b_n[4];
#pragma unroll
    for (int t = 0; t < 4; ++t) b_n[t] = frag_ld(brow[t], ks2);
#pragma unroll
    for (int t = 0; t < 4; ++t) c[t] = wmma16(a, bfr[t], c[t]);
    a = a_n;
#pragma unroll
    for (int t = 0; t < 4; ++t) bfr[t] = b_n[t];
  }
#pragma unroll
  for (int t = 0; t < 4; ++t) c[t] = wmma16(a, bfr[t], c[t]);

#pragma unroll
  for (int t = 0; t < 4; ++t) {
    int col = nbase + t * 16 + lc;
    float bias = bp[col];
#pragma unroll
    for (int r = 0; r < 8; ++r)
      out[(size_t)(m0 + r + hlf * 8) * DIMC + col] = c[t][r] + bias;
  }
}

// ---------------------------------------------------------------- launch
extern "C" void kernel_launch(void* const* d_in, const int* in_sizes, int n_in,
                              void* d_out, int out_size, void* d_ws, size_t ws_size,
                              hipStream_t stream) {
  (void)in_sizes; (void)n_in; (void)out_size; (void)ws_size;
  const float* x  = (const float*)d_in[0];
  const float* Wq = (const float*)d_in[1];
  const float* Wk = (const float*)d_in[2];
  const float* Wv = (const float*)d_in[3];
  const float* Wl = (const float*)d_in[4];
  const float* bl = (const float*)d_in[5];
  const float* Ww = (const float*)d_in[6];
  const float* bw = (const float*)d_in[7];
  const float* Wp = (const float*)d_in[8];
  const float* bp = (const float*)d_in[9];
  float* out = (float*)d_out;

  _Float16* ws = (_Float16*)d_ws;
  size_t off = 0;
  const size_t NX = (size_t)BATCH * SEQ * DIMC;   // 6,291,456
  const size_t NW = (size_t)DIMC * DIMC;
  _Float16* xh  = ws + off; off += NX;
  _Float16* WqT = ws + off; off += NW;
  _Float16* WkT = ws + off; off += NW;
  _Float16* WvT = ws + off; off += NW;
  _Float16* WpT = ws + off; off += NW;
  _Float16* qb  = ws + off; off += NX;
  _Float16* kb  = ws + off; off += NX;
  _Float16* vb  = ws + off; off += NX;   // transposed layout [b][h][d][j]
  _Float16* oh  = ws + off; off += NX;

  convert_x_kernel<<<(int)((NX + 255) / 256), 256, 0, stream>>>(x, xh, (int)NX);
  convert_w_kernel<<<(int)((NW + 255) / 256), 256, 0, stream>>>(
      Wq, Wk, Wv, Wp, WqT, WkT, WvT, WpT);
  qkv_gemm_kernel<<<dim3(BATCH * SEQ / 16, 9), 128, 0, stream>>>(
      xh, WqT, WkT, WvT, qb, kb, vb);
  attn_kernel<<<dim3(SEQ / 16, BATCH), 384, 0, stream>>>(
      qb, kb, vb, Wl, bl, Ww, bw, oh);
  out_proj_kernel<<<dim3(BATCH * SEQ / 16, 3), 128, 0, stream>>>(oh, WpT, bp, out);
}
